// Attention_39943195853118
// MI455X (gfx1250) — matrix operand order repaired and checked by reading, not compiled, not run on hardware
//
#include <hip/hip_runtime.h>
#include <hip/hip_bf16.h>

typedef __attribute__((ext_vector_type(16))) _Float16 v16h;
typedef __attribute__((ext_vector_type(8)))  _Float16 v8h;
typedef __attribute__((ext_vector_type(4)))  _Float16 v4h;
typedef __attribute__((ext_vector_type(8)))  float    v8f;
typedef __attribute__((ext_vector_type(4)))  float    f4;

#define HD   1024
#define SS   2048
#define BB   64
#define E2H  2048   // 2*H

#if defined(__HIP_DEVICE_COMPILE__) && __has_builtin(__builtin_amdgcn_sched_barrier)
#define SCHED_FENCE() __builtin_amdgcn_sched_barrier(0)
#else
#define SCHED_FENCE()
#endif

// ------------------------------------------------------------------
// branchless tanh: 1 - 2/(e^{2x}+1); exact limits at +-inf, NaN prop.
// ------------------------------------------------------------------
__device__ __forceinline__ float fast_tanhf(float x) {
  float e = __expf(x + x);
  return 1.f - 2.f / (e + 1.f);
}

// ------------------------------------------------------------------
// zero-init helper (energy accumulator + context output get atomics)
// ------------------------------------------------------------------
__global__ void zero_kernel(float* __restrict__ p, int n) {
  int i = blockIdx.x * 256 + threadIdx.x;
  if (i < n) p[i] = 0.f;
}

// ------------------------------------------------------------------
// dec_f[b,k] = sum_h dh[b,h] * W_h[h,k]   (tiny GEMM, L2-resident)
// ------------------------------------------------------------------
__global__ __launch_bounds__(256) void dec_feat_kernel(
    const float* __restrict__ dh, const float* __restrict__ Wh,
    float* __restrict__ dec_f) {
  __shared__ float sdh[HD];
  const int b = blockIdx.y;
  const int k = blockIdx.x * 256 + threadIdx.x;
  for (int i = threadIdx.x; i < HD; i += 256) sdh[i] = dh[b * HD + i];
  __syncthreads();
  float acc = 0.f;
#pragma unroll 4
  for (int h = 0; h < HD; ++h) acc = fmaf(sdh[h], Wh[(size_t)h * HD + k], acc);
  dec_f[b * HD + k] = acc;
}

// ------------------------------------------------------------------
// Fused: enc_f tile (WMMA f16, fp32 accum) -> tanh(dec_f+.)·v -> energy
// Workgroup: 256 thr (8 waves), 128 (b,s)-rows x 128 hidden cols,
// K = 2H = 2048 stepped by 32. Double-buffered LDS; pipeline order:
// barrier | load(global->reg) | WMMA | store(reg->LDS), pinned with
// sched_barrier so the barrier fence never waits on in-flight loads.
// ------------------------------------------------------------------
#define MT  128
#define NT  128
#define KT  32
#define LDA 40   // padded LDS row stride (halves); conflict-free b128 reads

__global__ __launch_bounds__(256) void energy_kernel(
    const float* __restrict__ enc,    // [B,S,2H]
    const float* __restrict__ Ws,     // [2H,H] row-major (row = K, col = N)
    const float* __restrict__ dec_f,  // [B,H]
    const float* __restrict__ vvec,   // [H]
    float* __restrict__ energy)       // [B,S] (atomic accum over N-chunks)
{
  __shared__ __align__(16) _Float16 Alds[2][MT * LDA];
  __shared__ __align__(16) _Float16 Blds[2][NT * LDA];

  const int b  = blockIdx.z;
  const int s0 = blockIdx.x * MT;
  const int c0 = blockIdx.y * NT;

  const int tid  = threadIdx.x;
  const int wave = tid >> 5;      // wave32
  const int lane = tid & 31;
  const int half = lane >> 4;
  const int l16  = lane & 15;
  const int m0   = wave * 16;     // this wave's 16 rows within tile

  const float* Abase = enc + ((size_t)b * SS + s0) * E2H;

  // in-flight K-slab held in VGPRs between load and LDS store
  f4 aR[4], bR[4];

  auto stage_load = [&](int k0) {   // issue global loads, no wait here
#pragma unroll
    for (int j = 0; j < 4; ++j) {
      int idx = tid + j * 256;            // 1024 float4 total
      int row = idx >> 3;
      int kc  = (idx & 7) << 2;
      aR[j] = *(const f4*)(Abase + (size_t)row * E2H + k0 + kc);
    }
#pragma unroll
    for (int j = 0; j < 4; ++j) {
      int idx  = tid + j * 256;
      int krow = idx >> 5;                // 0..31
      int nc   = (idx & 31) << 2;         // 0..124
      bR[j] = *(const f4*)(Ws + (size_t)(k0 + krow) * HD + c0 + nc);
    }
  };

  auto stage_store = [&](int buf) {  // cvt fp32->f16, write LDS
    _Float16* Ab = &Alds[buf][0];
    _Float16* Bb = &Blds[buf][0];
#pragma unroll
    for (int j = 0; j < 4; ++j) {
      int idx = tid + j * 256;
      int row = idx >> 3;
      int kc  = (idx & 7) << 2;
      v4h p;
      p[0] = (_Float16)aR[j].x; p[1] = (_Float16)aR[j].y;
      p[2] = (_Float16)aR[j].z; p[3] = (_Float16)aR[j].w;
      *(v4h*)&Ab[row * LDA + kc] = p;     // one ds_store_b64
    }
#pragma unroll
    for (int j = 0; j < 4; ++j) {
      int idx  = tid + j * 256;
      int krow = idx >> 5;
      int nc   = (idx & 31) << 2;
      Bb[(nc + 0) * LDA + krow] = (_Float16)bR[j].x;   // transpose to [n][k]
      Bb[(nc + 1) * LDA + krow] = (_Float16)bR[j].y;
      Bb[(nc + 2) * LDA + krow] = (_Float16)bR[j].z;
      Bb[(nc + 3) * LDA + krow] = (_Float16)bR[j].w;
    }
  };

  v8f acc[8] = {};
  const int NSTEP = E2H / KT;             // 64

  stage_load(0);
  stage_store(0);
  int cur = 0;
  for (int step = 0; step < NSTEP; ++step) {
    __syncthreads();                      // fence sees only prior ds_stores

    // Phase 1: issue next slab's global loads (regs only, no wait)
    if (step + 1 < NSTEP) stage_load((step + 1) * KT);
    SCHED_FENCE();

    // Phase 2: WMMA from buf[cur]
    const _Float16* Ab = &Alds[cur][0];
    const _Float16* Bb = &Blds[cur][0];

    // A fragment (ISA 16-bit A 16x32): halves get K 0-7/16-23 vs 8-15/24-31
    const int am = m0 + l16;
    v8h alo = *(const v8h*)&Ab[am * LDA + half * 8];
    v8h ahi = *(const v8h*)&Ab[am * LDA + 16 + half * 8];
    v16h afrag;
#pragma unroll
    for (int i = 0; i < 8; ++i) { afrag[i] = alo[i]; afrag[i + 8] = ahi[i]; }

    // B fragment (ISA B layout): halves get K 0-7/16-23 vs 8-15/24-31
    v8h blo[2], bhi[2];
    blo[0] = *(const v8h*)&Bb[l16 * LDA + half * 8];
    bhi[0] = *(const v8h*)&Bb[l16 * LDA + 16 + half * 8];
#pragma unroll
    for (int t = 0; t < 8; ++t) {
      if (t + 1 < 8) {                    // prefetch next frag pair
        const int bn = (t + 1) * 16 + l16;
        blo[(t + 1) & 1] = *(const v8h*)&Bb[bn * LDA + half * 8];
        bhi[(t + 1) & 1] = *(const v8h*)&Bb[bn * LDA + 16 + half * 8];
      }
      v16h bfrag;
#pragma unroll
      for (int i = 0; i < 8; ++i) {
        bfrag[i] = blo[t & 1][i]; bfrag[i + 8] = bhi[t & 1][i];
      }
      acc[t] = __builtin_amdgcn_wmma_f32_16x16x32_f16(
          false, afrag, false, bfrag, (short)0, acc[t], false, false);
    }
    SCHED_FENCE();

    // Phase 3: convert + store next slab (s_wait_loadcnt lands here)
    if (step + 1 < NSTEP) stage_store(cur ^ 1);
    cur ^= 1;
  }

  // ---- epilogue: energy partial = sum_col v[col]*tanh(acc + dec_f[col]) ----
  const float* decb = dec_f + b * HD;
  float dv[8], vw[8];
#pragma unroll
  for (int t = 0; t < 8; ++t) {
    const int col = c0 + t * 16 + l16;
    dv[t] = decb[col];
    vw[t] = vvec[col];
  }
  float esum[8] = {0.f, 0.f, 0.f, 0.f, 0.f, 0.f, 0.f, 0.f};
#pragma unroll
  for (int t = 0; t < 8; ++t)
#pragma unroll
    for (int r = 0; r < 8; ++r)
      esum[r] += vw[t] * fast_tanhf(acc[t][r] + dv[t]);

  // reduce the 16 column-lanes inside each half-wave
#pragma unroll
  for (int r = 0; r < 8; ++r) {
    float x = esum[r];
    x += __shfl_xor(x, 1, 32);
    x += __shfl_xor(x, 2, 32);
    x += __shfl_xor(x, 4, 32);
    x += __shfl_xor(x, 8, 32);
    esum[r] = x;
  }
  if (l16 == 0) {
    // C layout: VGPR r, lanes 0-15 -> M=r, lanes 16-31 -> M=8+r
#pragma unroll
    for (int r = 0; r < 8; ++r) {
      const int srow = s0 + m0 + half * 8 + r;
      atomicAdd(&energy[b * SS + srow], esum[r]);
    }
  }
}

// ------------------------------------------------------------------
// masked softmax over S per batch (one block per b)
// ------------------------------------------------------------------
__global__ __launch_bounds__(256) void softmax_kernel(
    const float* __restrict__ energy, const int* __restrict__ mask,
    float* __restrict__ attn) {
  __shared__ float red[256];
  const int b = blockIdx.x;
  float vals[8];
  float m = -3.4e38f;
#pragma unroll
  for (int j = 0; j < 8; ++j) {
    const int s = threadIdx.x + j * 256;
    float e = energy[b * SS + s];
    if (mask[b * SS + s] == 0) e = -1e10f;
    vals[j] = e;
    m = fmaxf(m, e);
  }
  red[threadIdx.x] = m; __syncthreads();
  for (int off = 128; off > 0; off >>= 1) {
    if (threadIdx.x < off)
      red[threadIdx.x] = fmaxf(red[threadIdx.x], red[threadIdx.x + off]);
    __syncthreads();
  }
  m = red[0]; __syncthreads();
  float sum = 0.f;
#pragma unroll
  for (int j = 0; j < 8; ++j) { vals[j] = __expf(vals[j] - m); sum += vals[j]; }
  red[threadIdx.x] = sum; __syncthreads();
  for (int off = 128; off > 0; off >>= 1) {
    if (threadIdx.x < off) red[threadIdx.x] += red[threadIdx.x + off];
    __syncthreads();
  }
  const float inv = 1.f / red[0];
#pragma unroll
  for (int j = 0; j < 8; ++j) {
    const int s = threadIdx.x + j * 256;
    attn[b * SS + s] = vals[j] * inv;
  }
}

// ------------------------------------------------------------------
// context[b,e] = sum_s attn[b,s]*enc[b,s,e]  (coalesced along e,
// s-chunked with fp32 atomics). Attention weights staged to LDS with
// the gfx1250 async memory->LDS path; streaming loads prefetched.
// ------------------------------------------------------------------
__global__ __launch_bounds__(256) void context_kernel(
    const float* __restrict__ enc, const float* __restrict__ attn,
    float* __restrict__ ctx) {
  __shared__ __align__(16) float w[256];
  const int b  = blockIdx.z;
  const int e  = blockIdx.x * 256 + threadIdx.x;
  const int s0 = blockIdx.y * 256;

#if defined(__HIP_DEVICE_COMPILE__) && defined(__gfx1250__)
  {
    // per-lane async copy: LDS[w + tid*4] = attn[b*S + s0 + tid]
    unsigned lds_addr = (unsigned)(uintptr_t)&w[threadIdx.x];
    const float* gptr = attn + b * SS + s0 + threadIdx.x;
    asm volatile("global_load_async_to_lds_b32 %0, %1, off"
                 :: "v"(lds_addr), "v"(gptr) : "memory");
    asm volatile("s_wait_asynccnt 0x0" ::: "memory");
  }
#else
  w[threadIdx.x] = attn[b * SS + s0 + threadIdx.x];
#endif
  __syncthreads();

  const float* base = enc + ((size_t)b * SS + s0) * E2H + e;
  float acc = 0.f;
#pragma unroll 4
  for (int s = 0; s < 256; ++s) {
    if (s + 16 < 256) __builtin_prefetch(base + (size_t)(s + 16) * E2H, 0, 0);
    acc = fmaf(w[s], base[(size_t)s * E2H], acc);
  }
  atomicAdd(&ctx[b * E2H + e], acc);
}

// ------------------------------------------------------------------
extern "C" void kernel_launch(void* const* d_in, const int* in_sizes, int n_in,
                              void* d_out, int out_size, void* d_ws, size_t ws_size,
                              hipStream_t stream) {
  const float* dh   = (const float*)d_in[0];  // [B,H]
  const float* enc  = (const float*)d_in[1];  // [B,S,2H]
  const int*   mask = (const int*)  d_in[2];  // [B,S]
  const float* Wh   = (const float*)d_in[3];  // [H,H]
  const float* Ws   = (const float*)d_in[4];  // [2H,H]
  const float* v    = (const float*)d_in[5];  // [H]

  float* ctx  = (float*)d_out;                        // [B,2H]
  float* attn = (float*)d_out + (size_t)BB * E2H;     // [B,S]

  float* dec_f  = (float*)d_ws;                       // B*H  floats
  float* energy = (float*)d_ws + (size_t)BB * HD;     // B*S  floats

  zero_kernel<<<(BB * SS + 255) / 256, 256, 0, stream>>>(energy, BB * SS);
  zero_kernel<<<(BB * E2H + 255) / 256, 256, 0, stream>>>(ctx, BB * E2H);

  dec_feat_kernel<<<dim3(HD / 256, BB), 256, 0, stream>>>(dh, Wh, dec_f);

  energy_kernel<<<dim3(SS / MT, HD / NT, BB), 256, 0, stream>>>(
      enc, Ws, dec_f, v, energy);

  softmax_kernel<<<BB, 256, 0, stream>>>(energy, mask, attn);

  context_kernel<<<dim3(E2H / 256, SS / 256, BB), 256, 0, stream>>>(
      enc, attn, ctx);
}
